// SlidingWindowBlock_29875792511778
// MI455X (gfx1250) — matrix operand order repaired and checked by reading, not compiled, not run on hardware
//
#include <hip/hip_runtime.h>
#include <hip/hip_bf16.h>

#define D_MODELc 1024
#define N_HEADSc 16
#define D_HEADc  64
#define WINDOWc  128
#define D_FFc    3072
#define SEQc     2048
#define BATCHc   2
#define ROWSc    (BATCHc * SEQc) /* 4096 */

typedef __attribute__((ext_vector_type(16))) _Float16 v16h;
typedef __attribute__((ext_vector_type(8)))  _Float16 v8h;
typedef __attribute__((ext_vector_type(8)))  float    v8f;
typedef __attribute__((ext_vector_type(4)))  unsigned int u32x4;
typedef __attribute__((ext_vector_type(8)))  int          i32x8;
typedef __attribute__((ext_vector_type(4)))  int          i32x4;

#if defined(__HIP_DEVICE_COMPILE__) && defined(__has_builtin)
#if __has_builtin(__builtin_amdgcn_tensor_load_to_lds) && \
    __has_builtin(__builtin_amdgcn_s_wait_tensorcnt)
#define USE_TDM 1
#endif
#endif
#ifndef USE_TDM
#define USE_TDM 0
#endif

static __device__ __forceinline__ v16h cat16(v8h lo, v8h hi) {
  return __builtin_shufflevector(lo, hi, 0,1,2,3,4,5,6,7,8,9,10,11,12,13,14,15);
}

// ---------------------------------------------------------------------------
// f32 -> f16 elementwise convert (weights)
// ---------------------------------------------------------------------------
__global__ __launch_bounds__(256) void cvt_f32_to_f16(const float* __restrict__ in,
                                                      _Float16* __restrict__ out, int n) {
  int i = blockIdx.x * 256 + threadIdx.x;
  if (i < n) out[i] = (_Float16)in[i];
}

// ---------------------------------------------------------------------------
// LayerNorm over D=1024, one block (256 thr) per row, f16 output
// ---------------------------------------------------------------------------
__global__ __launch_bounds__(256) void ln_kernel(const float* __restrict__ x,
                                                 const float* __restrict__ gam,
                                                 const float* __restrict__ bet,
                                                 _Float16* __restrict__ out) {
  int row = blockIdx.x;
  const float* xr = x + (size_t)row * D_MODELc;
  float v[4], s = 0.f, s2 = 0.f;
#pragma unroll
  for (int i = 0; i < 4; ++i) {
    v[i] = xr[threadIdx.x + i * 256];
    s += v[i];
    s2 += v[i] * v[i];
  }
#pragma unroll
  for (int o = 16; o; o >>= 1) {
    s  += __shfl_xor(s, o, 32);
    s2 += __shfl_xor(s2, o, 32);
  }
  __shared__ float red[16];
  int w = threadIdx.x >> 5, lane = threadIdx.x & 31;
  if (lane == 0) { red[w] = s; red[8 + w] = s2; }
  __syncthreads();
  float ts = 0.f, ts2 = 0.f;
#pragma unroll
  for (int i = 0; i < 8; ++i) { ts += red[i]; ts2 += red[8 + i]; }
  float mu  = ts * (1.f / D_MODELc);
  float var = ts2 * (1.f / D_MODELc) - mu * mu;
  float inv = rsqrtf(var + 1e-5f);
  _Float16* orow = out + (size_t)row * D_MODELc;
#pragma unroll
  for (int i = 0; i < 4; ++i) {
    int c = threadIdx.x + i * 256;
    orow[c] = (_Float16)((v[i] - mu) * inv * gam[c] + bet[c]);
  }
}

// ---------------------------------------------------------------------------
// WMMA GEMM with TDM-staged A tiles and double-buffered LDS pipeline.
//   MODE 0: outF = acc + bias
//   MODE 1: outF = acc + bias + res
//   MODE 2: outH = gelu(acc + bias)      (exact gelu, f16 out)
// Block tile 128x128, BK=32, 256 threads = 8 waves (2 M x 4 N), wave = 64x32.
// ---------------------------------------------------------------------------
#define BM 128
#define BN 128
#define BK 32
#define LDA 40 /* 64B row + 16B pad = 80B pitch (TDM pad: interval 16dw, amount 4dw) */
#define LDB 40

#if USE_TDM
// Issue a TDM 2D tile load: BMxBK f16 tile from row-major (rows x K) matrix,
// with hardware LDS padding producing an 80-byte row pitch (= LDA halfs).
static __device__ __forceinline__ void tdm_load_tile(const _Float16* gptr,
                                                     unsigned lds_off, int K) {
  unsigned long long ga = (unsigned long long)(uintptr_t)gptr;
  u32x4 g0;
  g0[0] = 1u;                                     // count=1 (valid user D#)
  g0[1] = lds_off;                                // LDS byte address
  g0[2] = (unsigned)(ga & 0xffffffffu);           // global_addr[31:0]
  g0[3] = (unsigned)((ga >> 32) & 0x01ffffffu) | (2u << 30); // ga[56:32] | type=2
  i32x8 g1;
  g1[0] = (int)((1u << 16)       // data_size = 2 bytes
              | (1u << 20)       // pad_enable
              | (3u << 22)       // pad_interval: 16 DWORDs (=64B) between pads
              | (3u << 25));     // pad_amount: 4 DWORDs (=16B)
  g1[1] = (int)(((unsigned)K & 0xffffu) << 16);   // tensor_dim0[15:0]
  g1[2] = (int)((unsigned)ROWSc << 16);           // dim0[31:16]=0 | tensor_dim1[15:0]
  g1[3] = (int)((unsigned)BK << 16);              // dim1[31:16]=0 | tile_dim0
  g1[4] = BM;                                     // tile_dim1 | tile_dim2=0
  g1[5] = K;                                      // tensor_dim0_stride[31:0]
  g1[6] = 0;                                      // stride0[47:32] | stride1[15:0]
  g1[7] = 0;
  i32x4 z4 = {0, 0, 0, 0};
  i32x8 z8 = {0, 0, 0, 0, 0, 0, 0, 0};
  __builtin_amdgcn_tensor_load_to_lds(g0, g1, z4, z4, z8, 0);
}
#endif

template <int MODE>
__global__ __launch_bounds__(256) void gemm_wmma(const _Float16* __restrict__ A,
                                                 const _Float16* __restrict__ Bw,
                                                 const float* __restrict__ bias,
                                                 const float* __restrict__ res,
                                                 float* __restrict__ outF,
                                                 _Float16* __restrict__ outH,
                                                 int M, int N, int K) {
  __shared__ _Float16 As[2][BM * LDA];
  __shared__ _Float16 Bs[2][BN * LDB]; // transposed: Bs[buf][n*LDB + k]

  int tid = threadIdx.x;
  int bm = blockIdx.y * BM;
  int bn = blockIdx.x * BN;
  int w = tid >> 5, lane = tid & 31;
  int wm = (w & 1) * 64;  // wave M offset in block tile
  int wn = (w >> 1) * 32; // wave N offset
  int lr = lane & 15;
  int lh = lane >> 4;

  v8f zero = {};
  v8f acc[4][2];
#pragma unroll
  for (int i = 0; i < 4; ++i)
#pragma unroll
    for (int j = 0; j < 2; ++j) acc[i][j] = zero;

  int arow = tid >> 1;       // 0..127
  int acol = (tid & 1) * 16; // 0 / 16
  int brow = tid >> 3;       // 0..31
  int bcol = (tid & 7) * 16; // 0..112

  const _Float16* Ag = A + (size_t)(bm + arow) * K + acol;
  const _Float16* Bg = Bw + (size_t)brow * N + bn + bcol;
  const int NT = K / BK;
  (void)Ag;

  // ---- prologue: stage tile 0 into buffer 0 ----
  {
    const v8h* bg = (const v8h*)Bg;
    v8h b0r = bg[0], b1r = bg[1];
#if USE_TDM
    if (w == 0) tdm_load_tile(A + (size_t)bm * K, (unsigned)(uintptr_t)&As[0][0], K);
#else
    const v8h* ag = (const v8h*)Ag;
    v8h a0r = ag[0], a1r = ag[1];
    *(v8h*)&As[0][arow * LDA + acol] = a0r;
    *(v8h*)&As[0][arow * LDA + acol + 8] = a1r;
#endif
#pragma unroll
    for (int i = 0; i < 8; ++i) Bs[0][(bcol + i) * LDB + brow] = b0r[i];
#pragma unroll
    for (int i = 0; i < 8; ++i) Bs[0][(bcol + 8 + i) * LDB + brow] = b1r[i];
  }

  for (int kt = 0; kt < NT; ++kt) {
    int cur = kt & 1, nxt = cur ^ 1;
#if USE_TDM
    if (w == 0) __builtin_amdgcn_s_wait_tensorcnt((short)0); // A[cur] landed
#endif
    __syncthreads(); // A[cur] + Bs[cur] visible to all waves

    bool pre = (kt + 1 < NT);
    v8h nb0 = {}, nb1 = {};
#if !USE_TDM
    v8h na0 = {}, na1 = {};
#endif
    if (pre) {
      int k1 = (kt + 1) * BK;
      const v8h* bg = (const v8h*)(Bg + (size_t)k1 * N);
      nb0 = bg[0];
      nb1 = bg[1];
#if USE_TDM
      if (w == 0)
        tdm_load_tile(A + (size_t)bm * K + k1, (unsigned)(uintptr_t)&As[nxt][0], K);
#else
      const v8h* ag = (const v8h*)(Ag + k1);
      na0 = ag[0];
      na1 = ag[1];
#endif
    }

    // ---- fragments + 8 WMMAs on buffer `cur` ----
    v16h afr[4], bfr[2];
#pragma unroll
    for (int i = 0; i < 4; ++i) {
      const _Float16* p = &As[cur][(wm + i * 16 + lr) * LDA + lh * 8];
      afr[i] = cat16(*(const v8h*)p, *(const v8h*)(p + 16));
    }
#pragma unroll
    for (int j = 0; j < 2; ++j) {
      const _Float16* p = &Bs[cur][(wn + j * 16 + lr) * LDB + lh * 8];
      bfr[j] = cat16(*(const v8h*)p, *(const v8h*)(p + 16));
    }
#pragma unroll
    for (int i = 0; i < 4; ++i)
#pragma unroll
      for (int j = 0; j < 2; ++j)
        acc[i][j] = __builtin_amdgcn_wmma_f32_16x16x32_f16(
            false, afr[i], false, bfr[j], (short)0, acc[i][j], false, false);

    // ---- stage tile kt+1 into buffer `nxt` ----
    if (pre) {
#if !USE_TDM
      *(v8h*)&As[nxt][arow * LDA + acol] = na0;
      *(v8h*)&As[nxt][arow * LDA + acol + 8] = na1;
#endif
#pragma unroll
      for (int i = 0; i < 8; ++i) Bs[nxt][(bcol + i) * LDB + brow] = nb0[i];
#pragma unroll
      for (int i = 0; i < 8; ++i) Bs[nxt][(bcol + 8 + i) * LDB + brow] = nb1[i];
    }
  }

  // ---- epilogue: VGPR r -> M = r + 8*lh, lane&15 -> N ----
#pragma unroll
  for (int i = 0; i < 4; ++i) {
#pragma unroll
    for (int j = 0; j < 2; ++j) {
      int gn = bn + wn + j * 16 + lr;
      float bv = bias[gn];
#pragma unroll
      for (int r = 0; r < 8; ++r) {
        int gm = bm + wm + i * 16 + lh * 8 + r;
        float vv = acc[i][j][r] + bv;
        if (MODE == 1) vv += res[(size_t)gm * N + gn];
        if (MODE == 2) {
          float gel = 0.5f * vv * (1.0f + erff(vv * 0.70710678118654752f));
          outH[(size_t)gm * N + gn] = (_Float16)gel;
        } else {
          outF[(size_t)gm * N + gn] = vv;
        }
      }
    }
  }
}

// ---------------------------------------------------------------------------
// Sliding-window causal attention. qkv: (ROWS, 3*D_MODEL) f32, laid out
// [q | k | v] per row, head h at offset h*64. One wave per query; 8 waves per
// block cover 8 consecutive s of one (b,h). ctx written as f16 (B,S,H,dh).
// ---------------------------------------------------------------------------
__global__ __launch_bounds__(256) void attn_kernel(const float* __restrict__ qkv,
                                                   _Float16* __restrict__ ctx) {
  __shared__ float sc[8][WINDOWc];
  int w = threadIdx.x >> 5, lane = threadIdx.x & 31;
  int bh = blockIdx.x >> 8; // 256 blocks per (b,h)
  int s = ((blockIdx.x & 255) << 3) + w;
  int b = bh >> 4, h = bh & 15;

  size_t rowq = (size_t)(b * SEQc + s);
  const float* qp = qkv + rowq * (3 * D_MODELc) + h * D_HEADc;
  float q0 = qp[2 * lane], q1 = qp[2 * lane + 1];

  int jstart = (s > WINDOWc - 1) ? s - (WINDOWc - 1) : 0;
  int nk = s - jstart + 1;

  for (int jj = 0; jj < nk; ++jj) {
    size_t rk = (size_t)(b * SEQc + jstart + jj);
    const float* kp = qkv + rk * (3 * D_MODELc) + D_MODELc + h * D_HEADc;
    float p = q0 * kp[2 * lane] + q1 * kp[2 * lane + 1];
#pragma unroll
    for (int o = 16; o; o >>= 1) p += __shfl_xor(p, o, 32);
    if (lane == 0) sc[w][jj] = p * 0.125f; // 1/sqrt(64)
  }
  __syncthreads();

  float m = -3.0e38f;
  for (int jj = lane; jj < nk; jj += 32) m = fmaxf(m, sc[w][jj]);
#pragma unroll
  for (int o = 16; o; o >>= 1) m = fmaxf(m, __shfl_xor(m, o, 32));
  float sum = 0.f;
  for (int jj = lane; jj < nk; jj += 32) {
    float e = __expf(sc[w][jj] - m);
    sc[w][jj] = e;
    sum += e;
  }
#pragma unroll
  for (int o = 16; o; o >>= 1) sum += __shfl_xor(sum, o, 32);
  float inv = 1.f / sum;
  __syncthreads();

  float c0 = 0.f, c1 = 0.f;
  for (int jj = 0; jj < nk; ++jj) {
    size_t rv = (size_t)(b * SEQc + jstart + jj);
    const float* vp = qkv + rv * (3 * D_MODELc) + 2 * D_MODELc + h * D_HEADc;
    float p = sc[w][jj];
    c0 += p * vp[2 * lane];
    c1 += p * vp[2 * lane + 1];
  }
  _Float16* cp = ctx + rowq * D_MODELc + h * D_HEADc + 2 * lane;
  cp[0] = (_Float16)(c0 * inv);
  cp[1] = (_Float16)(c1 * inv);
}

// ---------------------------------------------------------------------------
extern "C" void kernel_launch(void* const* d_in, const int* in_sizes, int n_in,
                              void* d_out, int out_size, void* d_ws, size_t ws_size,
                              hipStream_t stream) {
  (void)in_sizes; (void)n_in; (void)out_size; (void)ws_size;
  const float* x     = (const float*)d_in[0];
  const float* w_qkv = (const float*)d_in[1];
  const float* b_qkv = (const float*)d_in[2];
  const float* w_out = (const float*)d_in[3];
  const float* b_out = (const float*)d_in[4];
  const float* w_ff1 = (const float*)d_in[5];
  const float* b_ff1 = (const float*)d_in[6];
  const float* w_ff2 = (const float*)d_in[7];
  const float* b_ff2 = (const float*)d_in[8];
  const float* ln1_g = (const float*)d_in[9];
  const float* ln1_b = (const float*)d_in[10];
  const float* ln2_g = (const float*)d_in[11];
  const float* ln2_b = (const float*)d_in[12];
  float* out = (float*)d_out;

  char* p = (char*)d_ws;
  auto alloc = [&](size_t bytes) -> char* {
    char* r = p;
    p += (bytes + 255) & ~(size_t)255;
    return r;
  };
  _Float16* wqkv16 = (_Float16*)alloc((size_t)D_MODELc * 3 * D_MODELc * 2);
  _Float16* wout16 = (_Float16*)alloc((size_t)D_MODELc * D_MODELc * 2);
  _Float16* wff1_16 = (_Float16*)alloc((size_t)D_MODELc * D_FFc * 2);
  _Float16* wff2_16 = (_Float16*)alloc((size_t)D_FFc * D_MODELc * 2);
  _Float16* h1 = (_Float16*)alloc((size_t)ROWSc * D_MODELc * 2);
  float*    qkv = (float*)alloc((size_t)ROWSc * 3 * D_MODELc * 4);
  _Float16* ctx = (_Float16*)alloc((size_t)ROWSc * D_MODELc * 2);
  float*    y1 = (float*)alloc((size_t)ROWSc * D_MODELc * 4);
  _Float16* h2 = (_Float16*)alloc((size_t)ROWSc * D_MODELc * 2);
  _Float16* g16 = (_Float16*)alloc((size_t)ROWSc * D_FFc * 2);

  // weights -> f16
  int nq = D_MODELc * 3 * D_MODELc, no = D_MODELc * D_MODELc;
  cvt_f32_to_f16<<<(nq + 255) / 256, 256, 0, stream>>>(w_qkv, wqkv16, nq);
  cvt_f32_to_f16<<<(no + 255) / 256, 256, 0, stream>>>(w_out, wout16, no);
  cvt_f32_to_f16<<<(nq + 255) / 256, 256, 0, stream>>>(w_ff1, wff1_16, nq);
  cvt_f32_to_f16<<<(nq + 255) / 256, 256, 0, stream>>>(w_ff2, wff2_16, nq);

  // LN1
  ln_kernel<<<ROWSc, 256, 0, stream>>>(x, ln1_g, ln1_b, h1);
  // QKV projection
  gemm_wmma<0><<<dim3(3 * D_MODELc / BN, ROWSc / BM), 256, 0, stream>>>(
      h1, wqkv16, b_qkv, nullptr, qkv, nullptr, ROWSc, 3 * D_MODELc, D_MODELc);
  // Sliding-window attention
  attn_kernel<<<ROWSc * N_HEADSc / 8, 256, 0, stream>>>(qkv, ctx);
  // Out projection + residual: y1 = x + ctx @ w_out + b_out
  gemm_wmma<1><<<dim3(D_MODELc / BN, ROWSc / BM), 256, 0, stream>>>(
      ctx, wout16, b_out, x, y1, nullptr, ROWSc, D_MODELc, D_MODELc);
  // LN2
  ln_kernel<<<ROWSc, 256, 0, stream>>>(y1, ln2_g, ln2_b, h2);
  // FF1 + GELU
  gemm_wmma<2><<<dim3(D_FFc / BN, ROWSc / BM), 256, 0, stream>>>(
      h2, wff1_16, b_ff1, nullptr, nullptr, g16, ROWSc, D_FFc, D_MODELc);
  // FF2 + residual: out = y1 + g @ w_ff2 + b_ff2
  gemm_wmma<1><<<dim3(D_MODELc / BN, ROWSc / BM), 256, 0, stream>>>(
      g16, wff2_16, b_ff2, y1, out, nullptr, ROWSc, D_MODELc, D_FFc);
}
